// AdaptiveSplineLayer_40553081209131
// MI455X (gfx1250) — matrix operand order repaired and checked by reading, not compiled, not run on hardware
//
#include <hip/hip_runtime.h>
#include <cstdint>
#include <cstddef>

// ---------------------------------------------------------------------------
// AdaptiveSplineLayer for MI455X (gfx1250):
//   out = spline(x) @ proj_w.T + proj_b + x @ res_w.T
// Fused as one K=2048 GEMM:  [spline|x](16384x2048) @ [proj_w|res_w]^T + b
// Precision: split-bf16 (hi+lo) with f32 WMMA accumulation:
//   a*b ~= a_hi*b_hi + a_hi*b_lo + a_lo*b_hi  (3x v_wmma_f32_16x16x32_bf16)
// Data movement: global_load_async_to_lds_b128 (ASYNCcnt) double-buffered LDS.
// GEMM shape per block: 128 threads (4 wave32s), 128x64 tile, each wave owns
// 32 rows (2 M-subtiles) so every B fragment read from LDS feeds 2 A tiles
// -> 24 ds_load_b128 : 24 wmma per stage per wave (LDS no longer the bound).
// ---------------------------------------------------------------------------

typedef __bf16 bf16;
typedef __attribute__((ext_vector_type(8)))  bf16  bf16x8;
typedef __attribute__((ext_vector_type(16))) bf16  bf16x16;
typedef __attribute__((ext_vector_type(8)))  float f32x8;

#define M_DIM   16384
#define IN_DIM  1024
#define OUT_DIM 1024
#define K_TOT   2048   // IN (spline half) + IN (residual half)

// ---- CDNA5 async global->LDS copy (16B) + wait, via inline asm -------------
__device__ __forceinline__ void async_cp16(const bf16* g, bf16* l) {
  // VDST = VGPR holding LDS byte offset; VADDR = 64-bit global address; no saddr.
  unsigned loff = (unsigned)(uintptr_t)l;   // low 32 bits of flat shared addr = LDS offset
  asm volatile("global_load_async_to_lds_b128 %0, %1, off"
               :: "v"(loff), "v"(g) : "memory");
}
__device__ __forceinline__ void wait_async0() {
  asm volatile("s_wait_asynccnt 0x0" ::: "memory");
}

// ---- f32 -> bf16 hi/lo split ----------------------------------------------
__device__ __forceinline__ void split_bf16(float v, bf16& h, bf16& l) {
  h = (bf16)v;
  l = (bf16)(v - (float)h);
}

// ---------------------------------------------------------------------------
// Kernel 1: per-feature prep — sort 12 knots, sigmoid-mask coeffs.
// Table row (16 floats): [c0..c11, grid_min, scale=(K-1)/range, 0, 0]
// ---------------------------------------------------------------------------
__global__ void __launch_bounds__(256)
kan_prep_kernel(const float* __restrict__ grid, const float* __restrict__ coeffs,
                const float* __restrict__ alive, float* __restrict__ table) {
  int i = blockIdx.x * 256 + threadIdx.x;
  if (i >= IN_DIM) return;
  float g[12], c[12], a[12];
  #pragma unroll
  for (int j = 0; j < 12; ++j) {
    g[j] = grid[i * 12 + j];
    c[j] = coeffs[i * 12 + j];
    a[j] = alive[i * 12 + j];
  }
  // insertion sort by g, carrying (c, a)
  for (int j = 1; j < 12; ++j) {
    float gk = g[j], ck = c[j], ak = a[j];
    int p = j - 1;
    while (p >= 0 && g[p] > gk) {
      g[p + 1] = g[p]; c[p + 1] = c[p]; a[p + 1] = a[p]; --p;
    }
    g[p + 1] = gk; c[p + 1] = ck; a[p + 1] = ak;
  }
  float* T = table + i * 16;
  #pragma unroll
  for (int j = 0; j < 12; ++j)
    T[j] = c[j] * (1.0f / (1.0f + expf(-a[j])));
  T[12] = g[0];
  T[13] = 11.0f / fmaxf(g[11] - g[0], 1e-6f);
  T[14] = 0.0f; T[15] = 0.0f;
}

// ---------------------------------------------------------------------------
// Kernel 2: spline eval + hi/lo split of both spline(x) and x into
// concatenated A (row-major 16384 x 2048): cols [0,1024)=spline, [1024,2048)=x
// ---------------------------------------------------------------------------
__global__ void __launch_bounds__(256)
kan_spline_kernel(const float* __restrict__ x, const float* __restrict__ table,
                  bf16* __restrict__ Ahi, bf16* __restrict__ Alo) {
  size_t idx = (size_t)blockIdx.x * 256 + threadIdx.x;   // 0 .. 16M-1
  int i = (int)(idx & (IN_DIM - 1));
  size_t m = idx >> 10;
  float xv = x[idx];
  const float* T = table + i * 16;
  float xn = (xv - T[12]) * T[13];
  xn = fminf(fmaxf(xn, 0.0f), 11.0f);
  int il = (int)xn; if (il > 10) il = 10;
  float frac = xn - (float)il;
  float cl = T[il], cr = T[il + 1];
  float sp = fmaf(frac, cr - cl, cl);

  size_t base = m * (size_t)K_TOT;
  bf16 h, l;
  split_bf16(sp, h, l);
  Ahi[base + i] = h;            Alo[base + i] = l;
  split_bf16(xv, h, l);
  Ahi[base + IN_DIM + i] = h;   Alo[base + IN_DIM + i] = l;
}

// ---------------------------------------------------------------------------
// Kernel 3: weight concat + hi/lo split. W (1024 x 2048) row-major,
// W[n][k] = k<1024 ? proj_w[n][k] : res_w[n][k-1024]
// ---------------------------------------------------------------------------
__global__ void __launch_bounds__(256)
kan_wsplit_kernel(const float* __restrict__ pw, const float* __restrict__ rw,
                  bf16* __restrict__ Whi, bf16* __restrict__ Wlo) {
  size_t idx = (size_t)blockIdx.x * 256 + threadIdx.x;  // 0 .. 2M-1
  int k = (int)(idx & (K_TOT - 1));
  size_t n = idx >> 11;
  float v = (k < IN_DIM) ? pw[n * IN_DIM + k] : rw[n * IN_DIM + (k - IN_DIM)];
  bf16 h, l; split_bf16(v, h, l);
  Whi[idx] = h; Wlo[idx] = l;
}

// ---------------------------------------------------------------------------
// Kernel 4: GEMM. Block = 128 threads (4 wave32s) -> 128x64 output tile.
// Wave w: rows [32w, 32w+32) as 2 M-subtiles, 4 column sub-tiles of 16.
// K-chunk 32, double-buffered 48KB LDS filled by async-to-LDS DMA.
// ---------------------------------------------------------------------------
#define TILE_M 128
#define TILE_N 64
#define TILE_K 32
#define NTHREADS 128
#define A_EL   (TILE_M * TILE_K)              // 4096 bf16 per array
#define B_EL   (TILE_N * TILE_K)              // 2048 bf16 per array
#define BUF_EL (2 * A_EL + 2 * B_EL)          // 12288 bf16 = 24KB / buffer

__global__ void __launch_bounds__(NTHREADS)
kan_gemm_kernel(const bf16* __restrict__ Ahi, const bf16* __restrict__ Alo,
                const bf16* __restrict__ Whi, const bf16* __restrict__ Wlo,
                const float* __restrict__ bias, float* __restrict__ out) {
  __shared__ bf16 smem[2 * BUF_EL];           // 48KB, double buffered

  const int tid  = threadIdx.x;
  const int m0   = blockIdx.y * TILE_M;
  const int n0   = blockIdx.x * TILE_N;
  const int wave = tid >> 5;                  // 0..3
  const int lane = tid & 31;
  const int half = lane >> 4;                 // 0: lanes 0-15, 1: lanes 16-31
  const int lid  = lane & 15;

  auto stage = [&](bf16* buf, int k0) {
    // A hi/lo: 128x32 each -> 512 x 16B chunks -> 4 per thread per array
    #pragma unroll
    for (int j = 0; j < 4; ++j) {
      int q   = tid + NTHREADS * j;
      int row = q >> 2;
      int c8  = (q & 3) << 3;
      size_t go = (size_t)(m0 + row) * K_TOT + k0 + c8;
      async_cp16(Ahi + go, buf + row * TILE_K + c8);
      async_cp16(Alo + go, buf + A_EL + row * TILE_K + c8);
    }
    // B hi/lo: 64x32 each -> 256 chunks -> 2 per thread per array
    #pragma unroll
    for (int j = 0; j < 2; ++j) {
      int q  = tid + NTHREADS * j;
      int n  = q >> 2;
      int c8 = (q & 3) << 3;
      size_t go = (size_t)(n0 + n) * K_TOT + k0 + c8;
      async_cp16(Whi + go, buf + 2 * A_EL + n * TILE_K + c8);
      async_cp16(Wlo + go, buf + 2 * A_EL + B_EL + n * TILE_K + c8);
    }
  };

  f32x8 acc[2][4] = {};                       // [m-subtile][n-subtile]

  stage(smem, 0);
  wait_async0();
  __syncthreads();

  const int NSTAGE = K_TOT / TILE_K;          // 64
  for (int s = 0; s < NSTAGE; ++s) {
    bf16* cur = smem + (s & 1) * BUF_EL;
    if (s + 1 < NSTAGE) stage(smem + ((s + 1) & 1) * BUF_EL, (s + 1) * TILE_K);

    const bf16* sAhi = cur;
    const bf16* sAlo = cur + A_EL;
    const bf16* sBhi = cur + 2 * A_EL;
    const bf16* sBlo = cur + 2 * A_EL + B_EL;

    // A fragments (16x32 bf16), 2 M-subtiles:
    // lane half 0 -> K {0..15}, half 1 -> K {16..31}, M = lid
    bf16x16 a_hi[2], a_lo[2];
    #pragma unroll
    for (int u = 0; u < 2; ++u) {
      const int rA = (wave * 32 + u * 16 + lid) * TILE_K;
      bf16x8 h0 = *(const bf16x8*)&sAhi[rA + 16 * half];
      bf16x8 h1 = *(const bf16x8*)&sAhi[rA + 16 * half + 8];
      bf16x8 l0 = *(const bf16x8*)&sAlo[rA + 16 * half];
      bf16x8 l1 = *(const bf16x8*)&sAlo[rA + 16 * half + 8];
      a_hi[u] = __builtin_shufflevector(h0, h1, 0,1,2,3,4,5,6,7,8,9,10,11,12,13,14,15);
      a_lo[u] = __builtin_shufflevector(l0, l1, 0,1,2,3,4,5,6,7,8,9,10,11,12,13,14,15);
    }

    #pragma unroll
    for (int t = 0; t < 4; ++t) {
      // B fragment (32x16): lane holds N = lid, K = 16*half .. 16*half+15
      const int rB = (t * 16 + lid) * TILE_K + 16 * half;
      bf16x16 b_hi = *(const bf16x16*)&sBhi[rB];
      bf16x16 b_lo = *(const bf16x16*)&sBlo[rB];
      #pragma unroll
      for (int u = 0; u < 2; ++u) {
        acc[u][t] = __builtin_amdgcn_wmma_f32_16x16x32_bf16(false, a_hi[u], false, b_hi,
                                                            (short)0, acc[u][t], false, false);
        acc[u][t] = __builtin_amdgcn_wmma_f32_16x16x32_bf16(false, a_hi[u], false, b_lo,
                                                            (short)0, acc[u][t], false, false);
        acc[u][t] = __builtin_amdgcn_wmma_f32_16x16x32_bf16(false, a_lo[u], false, b_hi,
                                                            (short)0, acc[u][t], false, false);
      }
    }
    wait_async0();
    __syncthreads();
  }

  // Epilogue: D layout — VGPR j: row = j + 8*half, col = lid
  #pragma unroll
  for (int u = 0; u < 2; ++u) {
    #pragma unroll
    for (int t = 0; t < 4; ++t) {
      int col = n0 + t * 16 + lid;
      float b = bias[col];
      #pragma unroll
      for (int j = 0; j < 8; ++j) {
        int row = m0 + wave * 32 + u * 16 + j + 8 * half;
        out[(size_t)row * OUT_DIM + col] = acc[u][t][j] + b;
      }
    }
  }
}

// ---------------------------------------------------------------------------
// Launch. Inputs (setup_inputs order): x, grid, coeffs, knot_alive,
// proj_w, proj_b, res_w. Output: 16384x1024 f32.
// Workspace: table 64KB + Ahi/Alo 64MB each + Whi/Wlo 4MB each (~136MB).
// ---------------------------------------------------------------------------
extern "C" void kernel_launch(void* const* d_in, const int* in_sizes, int n_in,
                              void* d_out, int out_size, void* d_ws, size_t ws_size,
                              hipStream_t stream) {
  const float* x      = (const float*)d_in[0];
  const float* grid   = (const float*)d_in[1];
  const float* coeffs = (const float*)d_in[2];
  const float* alive  = (const float*)d_in[3];
  const float* proj_w = (const float*)d_in[4];
  const float* proj_b = (const float*)d_in[5];
  const float* res_w  = (const float*)d_in[6];
  float* out = (float*)d_out;

  char* w = (char*)d_ws;
  float* table = (float*)w;                 w += (size_t)IN_DIM * 16 * sizeof(float);
  bf16* Ahi = (bf16*)w;                     w += (size_t)M_DIM * K_TOT * sizeof(bf16);
  bf16* Alo = (bf16*)w;                     w += (size_t)M_DIM * K_TOT * sizeof(bf16);
  bf16* Whi = (bf16*)w;                     w += (size_t)OUT_DIM * K_TOT * sizeof(bf16);
  bf16* Wlo = (bf16*)w;

  kan_prep_kernel<<<(IN_DIM + 255) / 256, 256, 0, stream>>>(grid, coeffs, alive, table);
  kan_spline_kernel<<<(M_DIM * IN_DIM) / 256, 256, 0, stream>>>(x, table, Ahi, Alo);
  kan_wsplit_kernel<<<(OUT_DIM * K_TOT) / 256, 256, 0, stream>>>(proj_w, res_w, Whi, Wlo);

  dim3 gemm_grid(OUT_DIM / TILE_N, M_DIM / TILE_M);   // (16, 128)
  kan_gemm_kernel<<<gemm_grid, NTHREADS, 0, stream>>>(Ahi, Alo, Whi, Wlo, proj_b, out);
}
